// NanoTacotron_3590592659882
// MI455X (gfx1250) — matrix operand order repaired and checked by reading, not compiled, not run on hardware
//
#include <hip/hip_runtime.h>
#include <hip/hip_bf16.h>
#include <math.h>

typedef __attribute__((ext_vector_type(16))) _Float16 v16h;
typedef __attribute__((ext_vector_type(8)))  _Float16 v8h;
typedef __attribute__((ext_vector_type(8)))  float    v8f;

#define WMMA(a, b, c) __builtin_amdgcn_wmma_f32_16x16x32_f16(false, (a), false, (b), (short)0, (c), false, false)

__device__ __forceinline__ float sigf(float x) { return 1.0f / (1.0f + __expf(-x)); }

// ---------------------------------------------------------------------------
// Branch-free vectorized fragment loaders (fp32 memory -> f16 fragments).
// Contract: full tiles (M,N mult of 16; K mult of 32; rows 16B aligned).
// A 16x32 layout: lane<16: M=lane, K={0..7}U{16..23}; lane>=16: K={8..15}U{24..31}
// B 32x16 layout: lane<16: N=lane, K={0..7}U{16..23}; lane>=16: K={8..15}U{24..31}
// ---------------------------------------------------------------------------
__device__ __forceinline__ v16h cvt16(const float4 x0, const float4 x1,
                                      const float4 x2, const float4 x3) {
  v16h r;
  r[0] = (_Float16)x0.x;  r[1] = (_Float16)x0.y;  r[2] = (_Float16)x0.z;  r[3] = (_Float16)x0.w;
  r[4] = (_Float16)x1.x;  r[5] = (_Float16)x1.y;  r[6] = (_Float16)x1.z;  r[7] = (_Float16)x1.w;
  r[8] = (_Float16)x2.x;  r[9] = (_Float16)x2.y;  r[10] = (_Float16)x2.z; r[11] = (_Float16)x2.w;
  r[12] = (_Float16)x3.x; r[13] = (_Float16)x3.y; r[14] = (_Float16)x3.z; r[15] = (_Float16)x3.w;
  return r;
}

__device__ __forceinline__ v16h ldA_v(const float* __restrict__ P, int ldp,
                                      int m0, int k0, int lane) {
  int l = lane & 15, hi = lane >> 4;
  const float* row = P + (size_t)(m0 + l) * ldp + k0 + hi * 8;
  const float4* p = (const float4*)row;
  const float4* q = (const float4*)(row + 16);
  return cvt16(p[0], p[1], q[0], q[1]);
}

// A fragment from an explicit per-lane row pointer (channels-last conv)
__device__ __forceinline__ v16h ldA_row(const float* __restrict__ row, int k0, int lane) {
  int hi = lane >> 4;
  const float4* p = (const float4*)(row + k0 + hi * 8);
  const float4* q = (const float4*)(row + k0 + hi * 8 + 16);
  return cvt16(p[0], p[1], q[0], q[1]);
}

__device__ __forceinline__ v16h ldB_v(const float* __restrict__ Wm, int ldw,
                                      int n0, int k0, int lane) {
  int l = lane & 15, hi = lane >> 4;
  const float4* p = (const float4*)(Wm + (size_t)(n0 + l) * ldw + k0 + hi * 8);
  return cvt16(p[0], p[1], p[4], p[5]);
}

// A fragment from an f16 LDS tile: two ds_load_b128 per lane
__device__ __forceinline__ v16h ldA_lds(const _Float16* P, int ldp, int m0, int k0, int lane) {
  int l = lane & 15, hi = lane >> 4;
  const _Float16* row = P + (size_t)(m0 + l) * ldp + k0 + hi * 8;
  v8h lo = *(const v8h*)row;
  v8h hs = *(const v8h*)(row + 16);
  v16h r;
#pragma unroll
  for (int t = 0; t < 8; ++t) { r[t] = lo[t]; r[t + 8] = hs[t]; }
  return r;
}

// ---------------------------------------------------------------------------
// Generic GEMM: C[M,N] = A[M,K] @ W[N,K]^T + b1 + b2   (wave-per-16x16 tile)
// REQUIRES: M,N multiples of 16; K multiple of 32.
// ---------------------------------------------------------------------------
__global__ void __launch_bounds__(256) gemm_bias(const float* __restrict__ A,
                                                 const float* __restrict__ W,
                                                 const float* __restrict__ b1,
                                                 const float* __restrict__ b2,
                                                 float* __restrict__ C,
                                                 int M, int N, int K) {
  int lane = threadIdx.x & 31;
  int wave = blockIdx.x * (blockDim.x >> 5) + (threadIdx.x >> 5);
  int mt = M >> 4, nt = N >> 4;
  if (wave >= mt * nt) return;  // wave-uniform
  int i = wave / nt, j = wave % nt;
  v8f acc = {};
  for (int k0 = 0; k0 < K; k0 += 32) {
    v16h a = ldA_v(A, K, i * 16, k0, lane);
    v16h b = ldB_v(W, K, j * 16, k0, lane);
    if (k0 + 32 < K) {  // speculative prefetch of next chunk
      __builtin_prefetch(A + (size_t)(i * 16 + (lane & 15)) * K + k0 + 32, 0, 1);
      __builtin_prefetch(W + (size_t)(j * 16 + (lane & 15)) * K + k0 + 32, 0, 1);
    }
    acc = WMMA(a, b, acc);
  }
  int l = lane & 15, hi = lane >> 4;
  int n = j * 16 + l;
  float bias = (b1 ? b1[n] : 0.0f) + (b2 ? b2[n] : 0.0f);
#pragma unroll
  for (int r = 0; r < 8; ++r) {
    int m = i * 16 + hi * 8 + r;
    C[(size_t)m * N + n] = acc[r] + bias;
  }
}

// ---------------------------------------------------------------------------
// Channels-last conv1d as GEMM + fused BN + act.
// X:[B,S,Cip] (pad channels zeroed)  Wp:[KW,Co,Cip]  Y:[B,S,Co]
// REQUIRES: S,Co multiples of 16; Cip multiple of 32.
// A rows = output positions (channel-contiguous b128 loads); B = packed weights.
// Boundary taps: clamped-row loads + v_cndmask zeroing; no exec churn.
// ---------------------------------------------------------------------------
__global__ void __launch_bounds__(256) conv1d_cl(
    const float* __restrict__ X, const float* __restrict__ Wp,
    const float* __restrict__ bias, const float* __restrict__ g,
    const float* __restrict__ bb, const float* __restrict__ mn,
    const float* __restrict__ vr, float* __restrict__ Y,
    int Cip, int Co, int S, int KW, int act) {
  int b = blockIdx.z;
  const float* x = X + (size_t)b * S * Cip;
  float* y = Y + (size_t)b * S * Co;
  int lane = threadIdx.x & 31;
  int wave = blockIdx.x * (blockDim.x >> 5) + (threadIdx.x >> 5);
  int mt = S >> 4, nt = Co >> 4;
  if (wave >= mt * nt) return;
  int i = wave / nt, j = wave % nt;
  int pad = KW >> 1;
  int l = lane & 15, hi = lane >> 4;
  int srow = i * 16 + l;  // output position for this lane's A row
  v8f acc = {};
  for (int tap = 0; tap < KW; ++tap) {
    int si = srow + tap - pad;
    bool ok = (si >= 0) && (si < S);
    int sic = si < 0 ? 0 : (si >= S ? S - 1 : si);
    const float* arow = x + (size_t)sic * Cip;
    const float* wt = Wp + (size_t)tap * Co * Cip;
    for (int k0 = 0; k0 < Cip; k0 += 32) {
      v16h a = ldA_row(arow, k0, lane);
      if (!ok) {
#pragma unroll
        for (int t = 0; t < 16; ++t) a[t] = (_Float16)0.0f;
      }
      v16h bf = ldB_v(wt, Cip, j * 16, k0, lane);
      acc = WMMA(a, bf, acc);
    }
  }
  int n = j * 16 + l;  // output channel
  float bi = bias[n];
  float iv = rsqrtf(vr[n] + 1e-5f) * g[n];
  float mu = mn[n], be = bb[n];
#pragma unroll
  for (int r = 0; r < 8; ++r) {
    int m = i * 16 + hi * 8 + r;
    float tv = (acc[r] + bi - mu) * iv + be;
    if (act == 0) tv = fmaxf(tv, 0.0f);
    else if (act == 1) tv = tanhf(tv);
    y[(size_t)m * Co + n] = tv;
  }
}

// ---------------------------------------------------------------------------
// conv2d 3x3 stride2 pad1 (ref encoder, ~2 GFLOP total) — channel-major,
// clamp+select guards. X:[B,Ci,Hi,Wi] W:[Co,Ci,3,3] Y:[B,Co,Ho,Wo]
// ---------------------------------------------------------------------------
__global__ void __launch_bounds__(256) conv2d_bn_relu(
    const float* __restrict__ X, const float* __restrict__ W,
    const float* __restrict__ bias, const float* __restrict__ g,
    const float* __restrict__ bb, const float* __restrict__ mn,
    const float* __restrict__ vr, float* __restrict__ Y,
    int Ci, int Co, int Hi, int Wi, int Ho, int Wo) {
  int b = blockIdx.z;
  const float* x = X + (size_t)b * Ci * Hi * Wi;
  int NN = Ho * Wo;
  float* y = Y + (size_t)b * Co * NN;
  int lane = threadIdx.x & 31;
  int wave = blockIdx.x * (blockDim.x >> 5) + (threadIdx.x >> 5);
  int mt = (Co + 15) >> 4, nt = (NN + 15) >> 4;
  if (wave >= mt * nt) return;
  int i = wave / nt, j = wave % nt;
  int l = lane & 15, hi = lane >> 4;
  int mm = i * 16 + l;  // Co multiple of 16 for all layers
  int n = j * 16 + l;
  int nc = n < NN ? n : NN - 1;
  int oh = nc / Wo, ow = nc - oh * Wo;
  v8f acc = {};
  for (int kh = 0; kh < 3; ++kh) {
    for (int kw = 0; kw < 3; ++kw) {
      int ih = oh * 2 - 1 + kh, iw = ow * 2 - 1 + kw;
      bool ok = (n < NN) && (ih >= 0) && (ih < Hi) && (iw >= 0) && (iw < Wi);
      int ihc = ih < 0 ? 0 : (ih >= Hi ? Hi - 1 : ih);
      int iwc = iw < 0 ? 0 : (iw >= Wi ? Wi - 1 : iw);
      for (int k0 = 0; k0 < Ci; k0 += 32) {
        v16h a, bf;
#pragma unroll
        for (int t = 0; t < 8; ++t) {
          int k1 = k0 + hi * 8 + t, k2 = k1 + 16;
          int k1c = k1 < Ci ? k1 : Ci - 1, k2c = k2 < Ci ? k2 : Ci - 1;
          float w1 = W[((size_t)mm * Ci + k1c) * 9 + kh * 3 + kw];
          float w2 = W[((size_t)mm * Ci + k2c) * 9 + kh * 3 + kw];
          a[t]     = (k1 < Ci) ? (_Float16)w1 : (_Float16)0.0f;
          a[t + 8] = (k2 < Ci) ? (_Float16)w2 : (_Float16)0.0f;
        }
#pragma unroll
        for (int t = 0; t < 16; ++t) {
          int k = k0 + hi * 8 + (t & 7) + ((t >> 3) << 4);
          int kc = k < Ci ? k : Ci - 1;
          float xv = x[(size_t)kc * Hi * Wi + ihc * Wi + iwc];
          bf[t] = (ok && k < Ci) ? (_Float16)xv : (_Float16)0.0f;
        }
        acc = WMMA(a, bf, acc);
      }
    }
  }
  if (n < NN) {
#pragma unroll
    for (int r = 0; r < 8; ++r) {
      int co = i * 16 + hi * 8 + r;
      float t = acc[r] + bias[co];
      t = (t - mn[co]) * rsqrtf(vr[co] + 1e-5f) * g[co] + bb[co];
      y[(size_t)co * NN + n] = fmaxf(t, 0.0f);
    }
  }
}

// ---------------------------------------------------------------------------
// Persistent encoder LSTM: single WG, 8 waves, T sequential steps.
// Gx:[T*64,512] = x@Wih^T + biases (pre-GEMMed). Whh:[512,128]. Hs:[T,64,128]
// h f16 in LDS (ds_load_b128 A-fragments), c f32 in LDS, gate matrix in L2.
// ---------------------------------------------------------------------------
__global__ void __launch_bounds__(256) lstm_enc_persistent(
    const float* __restrict__ Gx, const float* __restrict__ Whh,
    float* __restrict__ Gs, float* __restrict__ Hs, int T, int rev) {
  __shared__ __align__(16) _Float16 hh[64 * 128];
  __shared__ float cs[64 * 128];
  int tid = threadIdx.x, lane = tid & 31, wv = tid >> 5;
  for (int idx = tid; idx < 64 * 128; idx += 256) { hh[idx] = (_Float16)0.0f; cs[idx] = 0.0f; }
  __syncthreads();
  for (int st = 0; st < T; ++st) {
    int t = rev ? (T - 1 - st) : st;
    const float* gx = Gx + (size_t)t * 64 * 512;
    for (int tile = wv; tile < 4 * 32; tile += 8) {
      int i = tile >> 5, j = tile & 31;
      v8f acc = {};
#pragma unroll
      for (int k0 = 0; k0 < 128; k0 += 32) {
        v16h a = ldA_lds(hh, 128, i * 16, k0, lane);
        v16h b = ldB_v(Whh, 128, j * 16, k0, lane);
        acc = WMMA(a, b, acc);
      }
      int l = lane & 15, hi = lane >> 4;
      int n = j * 16 + l;
#pragma unroll
      for (int r = 0; r < 8; ++r) {
        int m = i * 16 + hi * 8 + r;
        Gs[m * 512 + n] = acc[r] + gx[(size_t)m * 512 + n];
      }
    }
    __threadfence();
    __syncthreads();
    for (int idx = tid; idx < 64 * 128; idx += 256) {
      int b = idx >> 7, hc = idx & 127;
      const float* g0 = Gs + b * 512;
      float ig = g0[hc], fg = g0[hc + 128], gg = g0[hc + 256], og = g0[hc + 384];
      float cc = sigf(fg) * cs[idx] + sigf(ig) * tanhf(gg);
      cs[idx] = cc;
      float hv = sigf(og) * tanhf(cc);
      hh[idx] = (_Float16)hv;
      Hs[((size_t)t * 64 + b) * 128 + hc] = hv;
    }
    __threadfence();
    __syncthreads();
  }
}

// ---------------------------------------------------------------------------
// Persistent GRU (ref encoder): T' steps, Gi:[T*64,384] = x@Wih^T + bih
// ---------------------------------------------------------------------------
__global__ void __launch_bounds__(256) gru_persistent(
    const float* __restrict__ Gi, const float* __restrict__ Whh,
    const float* __restrict__ bhh, float* __restrict__ Gh,
    float* __restrict__ style, int T) {
  __shared__ float hs[64 * 128];
  __shared__ __align__(16) _Float16 hh[64 * 128];
  int tid = threadIdx.x, lane = tid & 31, wv = tid >> 5;
  for (int idx = tid; idx < 64 * 128; idx += 256) { hs[idx] = 0.0f; hh[idx] = (_Float16)0.0f; }
  __syncthreads();
  for (int t = 0; t < T; ++t) {
    for (int tile = wv; tile < 4 * 24; tile += 8) {
      int i = tile / 24, j = tile % 24;
      v8f acc = {};
#pragma unroll
      for (int k0 = 0; k0 < 128; k0 += 32) {
        v16h a = ldA_lds(hh, 128, i * 16, k0, lane);
        v16h b = ldB_v(Whh, 128, j * 16, k0, lane);
        acc = WMMA(a, b, acc);
      }
      int l = lane & 15, hi = lane >> 4;
      int n = j * 16 + l;
#pragma unroll
      for (int r = 0; r < 8; ++r) {
        int m = i * 16 + hi * 8 + r;
        Gh[m * 384 + n] = acc[r] + bhh[n];
      }
    }
    __threadfence();
    __syncthreads();
    for (int idx = tid; idx < 64 * 128; idx += 256) {
      int b = idx >> 7, hc = idx & 127;
      const float* gi = Gi + ((size_t)t * 64 + b) * 384;
      const float* gh = Gh + b * 384;
      float r = sigf(gi[hc] + gh[hc]);
      float z = sigf(gi[hc + 128] + gh[hc + 128]);
      float nn = tanhf(gi[hc + 256] + r * gh[hc + 256]);
      float hv = (1.0f - z) * nn + z * hs[idx];
      hs[idx] = hv;
      hh[idx] = (_Float16)hv;
    }
    __threadfence();
    __syncthreads();
  }
  for (int idx = tid; idx < 64 * 128; idx += 256) style[idx] = hs[idx];
}

// ---------------------------------------------------------------------------
// Persistent autoregressive decoder: one 1024-thread WG (32 waves), T steps.
// h1,h2 f32 in L2-resident scratch; c1,c2 in registers; li lda=480 zero-padded;
// Wih1 pre-padded to [1024,480]: every GEMM phase is unguarded + vectorized.
// ---------------------------------------------------------------------------
__global__ void __launch_bounds__(1024) decoder_persistent(
    const float* __restrict__ mel_inputs,  // [B,T,80]
    const float* __restrict__ enc_out,     // [B,S,384]
    const float* __restrict__ v_enc,       // [B,S,128]
    const float* __restrict__ Ww, const float* __restrict__ Wb,
    const float* __restrict__ uw, const float* __restrict__ ub,
    const float* __restrict__ W1p,  // padded Wih1 [1024,480]
    const float* __restrict__ Whh1,
    const float* __restrict__ bih1, const float* __restrict__ bhh1,
    const float* __restrict__ Wih2, const float* __restrict__ Whh2,
    const float* __restrict__ bih2, const float* __restrict__ bhh2,
    const float* __restrict__ melw, const float* __restrict__ melb,
    const float* __restrict__ gatew, const float* __restrict__ gateb,
    float* __restrict__ G, float* __restrict__ wh, float* __restrict__ sc,
    float* __restrict__ li, float* __restrict__ h1, float* __restrict__ h2,
    float* __restrict__ out_mel, float* __restrict__ out_gate,
    float* __restrict__ out_align, int T, int S) {
  const int tid = threadIdx.x, lane = tid & 31, wv = tid >> 5;
  float c1r[16], c2r[16];
#pragma unroll
  for (int u = 0; u < 16; ++u) { c1r[u] = 0.0f; c2r[u] = 0.0f; }
  for (int idx = tid; idx < 64 * 256; idx += 1024) { h1[idx] = 0.0f; h2[idx] = 0.0f; }
  for (int idx = tid; idx < 64 * 480; idx += 1024) li[idx] = 0.0f;  // zero incl. pad tail
  __threadfence();
  __syncthreads();

  for (int t = 0; t < T; ++t) {
    // ---- phase 1: wh = h2 @ Ww^T + Wb  [64,128] K=256; 32 tiles, 1/wave
    {
      int i = wv >> 3, j = wv & 7;
      v8f acc = {};
#pragma unroll
      for (int k0 = 0; k0 < 256; k0 += 32) {
        v16h a = ldA_v(h2, 256, i * 16, k0, lane);
        v16h b = ldB_v(Ww, 256, j * 16, k0, lane);
        acc = WMMA(a, b, acc);
      }
      int l = lane & 15, hi = lane >> 4;
      int n = j * 16 + l;
#pragma unroll
      for (int r = 0; r < 8; ++r) wh[(i * 16 + hi * 8 + r) * 128 + n] = acc[r] + Wb[n];
    }
    __threadfence();
    __syncthreads();
    // ---- phase 2: scores[b,s] = u . tanh(wh[b] + v_enc[b,s]) + ub
    for (int idx = tid; idx < 64 * S; idx += 1024) {
      int b = idx / S, s = idx - b * S;
      const float* ve = v_enc + ((size_t)b * S + s) * 128;
      const float* w0 = wh + b * 128;
      float e = 0.0f;
      for (int a = 0; a < 128; ++a) e += tanhf(w0[a] + ve[a]) * uw[a];
      sc[idx] = e + ub[0];
    }
    __threadfence();
    __syncthreads();
    // ---- phase 3: per-row softmax + emit alignments
    if (tid < 64) {
      float* row = sc + tid * S;
      float mx = -1e30f;
      for (int s = 0; s < S; ++s) mx = fmaxf(mx, row[s]);
      float sum = 0.0f;
      for (int s = 0; s < S; ++s) { float e = __expf(row[s] - mx); row[s] = e; sum += e; }
      float inv = 1.0f / sum;
      float* al = out_align + ((size_t)tid * T + t) * S;
      for (int s = 0; s < S; ++s) { float w = row[s] * inv; row[s] = w; al[s] = w; }
    }
    __threadfence();
    __syncthreads();
    // ---- phase 4: li = [context(384), mel_in(80)], lda=480 (tail stays 0)
    for (int idx = tid; idx < 64 * 464; idx += 1024) {
      int b = idx / 464, d = idx - b * 464;
      float v;
      if (d < 384) {
        const float* wrow = sc + b * S;
        const float* eo = enc_out + (size_t)b * S * 384 + d;
        float a2 = 0.0f;
        for (int s = 0; s < S; ++s) a2 += wrow[s] * eo[(size_t)s * 384];
        v = a2;
      } else {
        v = (t == 0) ? 0.0f : mel_inputs[((size_t)b * T + (t - 1)) * 80 + (d - 384)];
      }
      li[b * 480 + d] = v;
    }
    __threadfence();
    __syncthreads();
    // ---- phase 5: G = li@W1p^T + h1@Whh1^T   [64,1024]
    for (int tile = wv; tile < 4 * 64; tile += 32) {
      int i = tile >> 6, j = tile & 63;
      v8f acc = {};
      for (int k0 = 0; k0 < 480; k0 += 32) {
        v16h a = ldA_v(li, 480, i * 16, k0, lane);
        v16h b = ldB_v(W1p, 480, j * 16, k0, lane);
        acc = WMMA(a, b, acc);
      }
#pragma unroll
      for (int k0 = 0; k0 < 256; k0 += 32) {
        v16h a = ldA_v(h1, 256, i * 16, k0, lane);
        v16h b = ldB_v(Whh1, 256, j * 16, k0, lane);
        acc = WMMA(a, b, acc);
      }
      int l = lane & 15, hi = lane >> 4;
      int n = j * 16 + l;
#pragma unroll
      for (int r = 0; r < 8; ++r) G[(i * 16 + hi * 8 + r) * 1024 + n] = acc[r];
    }
    __threadfence();
    __syncthreads();
    // ---- phase 6: lstm1 gate math (c1 in regs, h1 -> scratch)
#pragma unroll
    for (int u = 0; u < 16; ++u) {
      int idx = u * 1024 + tid;
      int b = idx >> 8, hc = idx & 255;
      const float* g0 = G + b * 1024;
      float ig = g0[hc] + bih1[hc] + bhh1[hc];
      float fg = g0[hc + 256] + bih1[hc + 256] + bhh1[hc + 256];
      float gg = g0[hc + 512] + bih1[hc + 512] + bhh1[hc + 512];
      float og = g0[hc + 768] + bih1[hc + 768] + bhh1[hc + 768];
      float cc = sigf(fg) * c1r[u] + sigf(ig) * tanhf(gg);
      c1r[u] = cc;
      h1[idx] = sigf(og) * tanhf(cc);
    }
    __threadfence();
    __syncthreads();
    // ---- phase 7: G = h1@Wih2^T + h2@Whh2^T
    for (int tile = wv; tile < 4 * 64; tile += 32) {
      int i = tile >> 6, j = tile & 63;
      v8f acc = {};
#pragma unroll
      for (int k0 = 0; k0 < 256; k0 += 32) {
        v16h a = ldA_v(h1, 256, i * 16, k0, lane);
        v16h b = ldB_v(Wih2, 256, j * 16, k0, lane);
        acc = WMMA(a, b, acc);
        v16h a2 = ldA_v(h2, 256, i * 16, k0, lane);
        v16h b2 = ldB_v(Whh2, 256, j * 16, k0, lane);
        acc = WMMA(a2, b2, acc);
      }
      int l = lane & 15, hi = lane >> 4;
      int n = j * 16 + l;
#pragma unroll
      for (int r = 0; r < 8; ++r) G[(i * 16 + hi * 8 + r) * 1024 + n] = acc[r];
    }
    __threadfence();
    __syncthreads();
    // ---- phase 8: lstm2 gate math -> h2
#pragma unroll
    for (int u = 0; u < 16; ++u) {
      int idx = u * 1024 + tid;
      int b = idx >> 8, hc = idx & 255;
      const float* g0 = G + b * 1024;
      float ig = g0[hc] + bih2[hc] + bhh2[hc];
      float fg = g0[hc + 256] + bih2[hc + 256] + bhh2[hc + 256];
      float gg = g0[hc + 512] + bih2[hc + 512] + bhh2[hc + 512];
      float og = g0[hc + 768] + bih2[hc + 768] + bhh2[hc + 768];
      float cc = sigf(fg) * c2r[u] + sigf(ig) * tanhf(gg);
      c2r[u] = cc;
      h2[idx] = sigf(og) * tanhf(cc);
    }
    __threadfence();
    __syncthreads();
    // ---- phase 9: mel projection (waves 0..19) + stop gate (waves 20..21)
    if (wv < 20) {
      int i = wv / 5, j = wv % 5;
      v8f acc = {};
#pragma unroll
      for (int k0 = 0; k0 < 256; k0 += 32) {
        v16h a = ldA_v(h2, 256, i * 16, k0, lane);
        v16h b = ldB_v(melw, 256, j * 16, k0, lane);
        acc = WMMA(a, b, acc);
      }
      int l = lane & 15, hi = lane >> 4;
      int n = j * 16 + l;  // < 80 always (N = 5*16)
#pragma unroll
      for (int r = 0; r < 8; ++r) {
        int m = i * 16 + hi * 8 + r;
        out_mel[((size_t)m * T + t) * 80 + n] = acc[r] + melb[n];
      }
    } else if (wv == 20 || wv == 21) {
      int b = tid - 640;
      const float4* hrow = (const float4*)(h2 + b * 256);
      const float4* gw = (const float4*)gatew;
      float a2 = gateb[0];
      for (int k = 0; k < 64; ++k) {
        float4 hv = hrow[k], wv4 = gw[k];
        a2 += hv.x * wv4.x + hv.y * wv4.y + hv.z * wv4.z + hv.w * wv4.w;
      }
      out_gate[(size_t)b * T + t] = sigf(a2);
    }
    __threadfence();
    __syncthreads();
  }
}

// ---------------------------------------------------------------------------
// Small elementwise / layout kernels
// ---------------------------------------------------------------------------
__global__ void k_embed(const int* __restrict__ idx, const float* __restrict__ emb,
                        float* __restrict__ x, int n) {  // x:[B,S,256] channels-last
  int i = blockIdx.x * blockDim.x + threadIdx.x;
  if (i >= n) return;
  int b = i >> 17, r = i & 131071;
  int s = r >> 8, c = r & 255;
  x[i] = emb[(size_t)idx[b * 512 + s] * 256 + c];
}

__global__ void k_xs_tr(const float* __restrict__ src, float* __restrict__ dst, int n) {
  // [B,S,256] -> rows (s*64+b) x 256
  int i = blockIdx.x * blockDim.x + threadIdx.x;
  if (i >= n) return;
  int s = i / (64 * 256), r = i - s * 64 * 256;
  int b = r >> 8, c = r & 255;
  dst[i] = src[((size_t)b * 512 + s) * 256 + c];
}

__global__ void k_wpack1d(const float* __restrict__ W, float* __restrict__ Wp,
                          int Co, int Ci, int Cip, int KW, int n) {
  // W:[Co,Ci,KW] -> Wp:[KW,Co,Cip] zero-padded channels
  int i = blockIdx.x * blockDim.x + threadIdx.x;
  if (i >= n) return;
  int tap = i / (Co * Cip), r = i - tap * Co * Cip;
  int co = r / Cip, c = r - co * Cip;
  Wp[i] = (c < Ci) ? W[((size_t)co * Ci + c) * KW + tap] : 0.0f;
}

__global__ void k_ref_tr(const float* __restrict__ src, float* __restrict__ dst, int n) {
  // ref_mel [B,200,80] -> [B,1,80,200]
  int i = blockIdx.x * blockDim.x + threadIdx.x;
  if (i >= n) return;
  int b = i / 16000, r = i - b * 16000;
  int f = r / 200, t = r - f * 200;
  dst[i] = src[(size_t)b * 16000 + t * 80 + f];
}

__global__ void k_rgru_tr(const float* __restrict__ src, float* __restrict__ dst, int n) {
  // [B,128,2,4] -> rows (t*64+b) x 256 with col = c*2+f
  int i = blockIdx.x * blockDim.x + threadIdx.x;
  if (i >= n) return;
  int row = i >> 8, c2 = i & 255;
  int t = row >> 6, b = row & 63;
  int c = c2 >> 1, f = c2 & 1;
  dst[i] = src[(size_t)b * 1024 + c * 8 + f * 4 + t];
}

__global__ void k_concat(const float* __restrict__ hf, const float* __restrict__ hb,
                         const float* __restrict__ style, float* __restrict__ eo, int n) {
  // enc_out [B,S,384]
  int i = blockIdx.x * blockDim.x + threadIdx.x;
  if (i >= n) return;
  int b = i / (512 * 384), r = i - b * 512 * 384;
  int s = r / 384, d = r - s * 384;
  float v;
  if (d < 128) v = hf[((size_t)s * 64 + b) * 128 + d];
  else if (d < 256) v = hb[((size_t)s * 64 + b) * 128 + (d - 128)];
  else v = style[b * 128 + (d - 256)];
  eo[i] = v;
}

__global__ void k_pad_w1(const float* __restrict__ w, float* __restrict__ dst, int n) {
  // Wih1 [1024,464] -> [1024,480] zero-padded
  int i = blockIdx.x * blockDim.x + threadIdx.x;
  if (i >= n) return;
  int r = i / 480, k = i - r * 480;
  dst[i] = (k < 464) ? w[(size_t)r * 464 + k] : 0.0f;
}

__global__ void k_pad_mel(const float* __restrict__ mel, float* __restrict__ dst, int n) {
  // [B*T,80] -> [B*T,96] zero-padded channels
  int i = blockIdx.x * blockDim.x + threadIdx.x;
  if (i >= n) return;
  int r = i / 96, c = i - r * 96;
  dst[i] = (c < 80) ? mel[(size_t)r * 80 + c] : 0.0f;
}

__global__ void k_final_add(const float* __restrict__ mel, const float* __restrict__ pn,
                            float* __restrict__ outF, int n) {
  // all [B,T,80] channels-last: plain elementwise add
  int i = blockIdx.x * blockDim.x + threadIdx.x;
  if (i >= n) return;
  outF[i] = mel[i] + pn[i];
}

// ---------------------------------------------------------------------------
// Host orchestration
// ---------------------------------------------------------------------------
static inline void gemmL(const float* A, const float* W, const float* b1, const float* b2,
                         float* C, int M, int N, int K, hipStream_t s) {
  int tiles = (M / 16) * (N / 16);
  gemm_bias<<<dim3((tiles + 7) / 8), dim3(256), 0, s>>>(A, W, b1, b2, C, M, N, K);
}

extern "C" void kernel_launch(void* const* d_in, const int* in_sizes, int n_in,
                              void* d_out, int out_size, void* d_ws, size_t ws_size,
                              hipStream_t stream) {
  (void)in_sizes; (void)out_size; (void)ws_size;
  const float* F[128];
  for (int i = 0; i < n_in && i < 128; ++i) F[i] = (const float*)d_in[i];
  const int* text = (const int*)d_in[0];
  const float* mel_in = F[1];
  const float* ref_mel = F[2];
  const float* emb = F[3];
  float* out = (float*)d_out;
  float* ws = (float*)d_ws;

  // workspace layout (floats)
  const size_t oPA = 0;                   // ping A     26,214,400
  const size_t oPB = oPA + 26214400;      // ping B     26,214,400
  const size_t oENC = oPB + 26214400;     // enc_out    12,582,912
  const size_t oVE = oENC + 12582912;     // v_enc       4,194,304
  const size_t oHF = oVE + 4194304;       // hs_f        4,194,304
  const size_t oHB = oHF + 4194304;       // hs_b        4,194,304
  const size_t oGX = oHB + 4194304;       // Gx         16,777,216
  const size_t oST = oGX + 16777216;      // style           8,192
  const size_t oGI = oST + 8192;          // gi_gru         98,304
  const size_t oG  = oGI + 98304;         // gates          65,536
  const size_t oWH = oG + 65536;          // wh              8,192
  const size_t oSC = oWH + 8192;          // scores         32,768
  const size_t oLI = oSC + 32768;         // li [64,480]    30,720
  const size_t oH1 = oLI + 30720;         // h1             16,384
  const size_t oH2 = oH1 + 16384;         // h2             16,384
  const size_t oGH = oH2 + 16384;         // gh_gru         24,576
  const size_t oW1P = oGH + 24576;        // Wih1 padded   491,520
  const size_t oWP = oW1P + 491520;       // packed conv W 1,310,720
  const size_t oMP = oWP + 1310720;       // padded mel   4,915,200

  float* pA = ws + oPA; float* pB = ws + oPB;
  float* Wp = ws + oWP;
  const int B = 64, S = 512, T = 800;

  // ---- 1. embedding -> [B,S,256] channels-last in pA
  { int n = B * S * 256; k_embed<<<dim3((n + 255) / 256), dim3(256), 0, stream>>>(text, emb, pA, n); }

  // ---- 2. encoder convs (channels-last, vectorized) A->B->A->B
  {
    float* src = pA; float* dst = pB;
    for (int l = 0; l < 3; ++l) {
      int base = 4 + l * 6;
      int np = 5 * 256 * 256;
      k_wpack1d<<<dim3((np + 255) / 256), dim3(256), 0, stream>>>(F[base], Wp, 256, 256, 256, 5, np);
      int tiles = (S / 16) * (256 / 16);
      conv1d_cl<<<dim3((tiles + 7) / 8, 1, B), dim3(256), 0, stream>>>(
          src, Wp, F[base + 1], F[base + 2], F[base + 3], F[base + 4], F[base + 5],
          dst, 256, 256, S, 5, 0);
      float* tmp = src; src = dst; dst = tmp;
    }
  }  // result [B,S,256] in pB

  // ---- 3. [B,S,256] -> xs rows (s*64+b)x256 in pA
  { int n = S * B * 256; k_xs_tr<<<dim3((n + 255) / 256), dim3(256), 0, stream>>>(pB, pA, n); }

  // ---- 4. BiLSTM: Gx GEMM + persistent scan, fwd then bwd (shared Gx buffer)
  gemmL(pA, F[22], F[24], F[25], ws + oGX, S * B, 512, 256, stream);
  lstm_enc_persistent<<<dim3(1), dim3(256), 0, stream>>>(ws + oGX, F[23], ws + oG, ws + oHF, S, 0);
  gemmL(pA, F[26], F[28], F[29], ws + oGX, S * B, 512, 256, stream);
  lstm_enc_persistent<<<dim3(1), dim3(256), 0, stream>>>(ws + oGX, F[27], ws + oG, ws + oHB, S, 1);

  // ---- 5. reference encoder: transpose + 6 conv2d + GRU
  { int n = B * 80 * 200; k_ref_tr<<<dim3((n + 255) / 256), dim3(256), 0, stream>>>(ref_mel, pB, n); }
  {
    const int Ci[6] = {1, 32, 32, 64, 64, 128}, Co[6] = {32, 32, 64, 64, 128, 128};
    const int Hi[6] = {80, 40, 20, 10, 5, 3}, Wi[6] = {200, 100, 50, 25, 13, 7};
    const int Ho[6] = {40, 20, 10, 5, 3, 2}, Wo[6] = {100, 50, 25, 13, 7, 4};
    float* src = pB; float* dst = pA;
    for (int l = 0; l < 6; ++l) {
      int base = 30 + l * 6;
      int tiles = ((Co[l] + 15) / 16) * ((Ho[l] * Wo[l] + 15) / 16);
      conv2d_bn_relu<<<dim3((tiles + 7) / 8, 1, B), dim3(256), 0, stream>>>(
          src, F[base], F[base + 1], F[base + 2], F[base + 3], F[base + 4], F[base + 5],
          dst, Ci[l], Co[l], Hi[l], Wi[l], Ho[l], Wo[l]);
      float* tmp = src; src = dst; dst = tmp;
    }
  }  // result [B,128,2,4] in pB
  { int n = 4 * B * 256; k_rgru_tr<<<dim3((n + 255) / 256), dim3(256), 0, stream>>>(pB, pA, n); }
  gemmL(pA, F[66], F[68], nullptr, ws + oGI, 4 * B, 384, 256, stream);
  gru_persistent<<<dim3(1), dim3(256), 0, stream>>>(ws + oGI, F[67], F[69], ws + oGH, ws + oST, 4);

  // ---- 6. enc_out concat + v_enc precompute + padded Wih1
  { int n = B * S * 384; k_concat<<<dim3((n + 255) / 256), dim3(256), 0, stream>>>(ws + oHF, ws + oHB, ws + oST, ws + oENC, n); }
  gemmL(ws + oENC, F[72], F[73], nullptr, ws + oVE, B * S, 128, 384, stream);
  { int n = 1024 * 480; k_pad_w1<<<dim3((n + 255) / 256), dim3(256), 0, stream>>>(F[76], ws + oW1P, n); }

  // ---- 7. autoregressive decoder (persistent, 800 steps)
  float* out_mel = out;
  float* out_final = out + 4096000;
  float* out_gate = out + 8192000;
  float* out_align = out + 8243200;
  decoder_persistent<<<dim3(1), dim3(1024), 0, stream>>>(
      mel_in, ws + oENC, ws + oVE,
      F[70], F[71], F[74], F[75],
      ws + oW1P, F[77], F[78], F[79],
      F[80], F[81], F[82], F[83],
      F[84], F[85], F[86], F[87],
      ws + oG, ws + oWH, ws + oSC, ws + oLI, ws + oH1, ws + oH2,
      out_mel, out_gate, out_align, T, S);

  // ---- 8. postnet (channels-last throughout; mel_outputs is already [B,T,80])
  { int n = B * T * 96; k_pad_mel<<<dim3((n + 255) / 256), dim3(256), 0, stream>>>(out_mel, ws + oMP, n); }
  {
    const int Ci[5] = {80, 512, 512, 512, 512}, Cip[5] = {96, 512, 512, 512, 512};
    const int Co[5] = {512, 512, 512, 512, 80};
    float* src = ws + oMP; float* dst = pA;
    for (int l = 0; l < 5; ++l) {
      int base = 88 + l * 6;
      int act = (l < 4) ? 1 : 2;  // tanh on first 4, none on last
      int np = 5 * Co[l] * Cip[l];
      k_wpack1d<<<dim3((np + 255) / 256), dim3(256), 0, stream>>>(F[base], Wp, Co[l], Ci[l], Cip[l], 5, np);
      int tiles = (T / 16) * (Co[l] / 16);
      conv1d_cl<<<dim3((tiles + 7) / 8, 1, B), dim3(256), 0, stream>>>(
          src, Wp, F[base + 1], F[base + 2], F[base + 3], F[base + 4], F[base + 5],
          dst, Cip[l], Co[l], T, 5, act);
      src = dst;
      dst = (dst == pA) ? pB : pA;
    }
  }  // result [B,T,80] in pA (after 5 layers: oMP->pA->pB->pA->pB->pA)
  { int n = B * T * 80; k_final_add<<<dim3((n + 255) / 256), dim3(256), 0, stream>>>(out_mel, pA, out_final, n); }
}
